// LlamaEngram_41472204210335
// MI455X (gfx1250) — matrix operand order repaired and checked by reading, not compiled, not run on hardware
//
#include <hip/hip_runtime.h>
#include <stdint.h>

typedef __bf16 bf16;
typedef __attribute__((ext_vector_type(16))) __bf16 v16bf;
typedef __attribute__((ext_vector_type(8)))  float  v8f;
typedef __attribute__((ext_vector_type(4)))  int    v4i;

union FragAB { v16bf v; uint4 q[2]; };

#define D_HID   4096
#define E_DIM   1024
#define T_SEQ   4096
#define B_BATCH 2
#define NTOK    (B_BATCH * T_SEQ)   // 8192
#define VOCAB   200000
#define HEAD_DIM 64
#define EPS_F   1e-5f
#define MOUT    32                  // output tokens per fused value/conv block
#define MTILE   48                  // 16-row halo tile + 32 output rows
#define NCHUNKS (D_HID / 16)        // 256
#define ITERS   (NCHUNKS / 8)       // 32 chunks per wave

// ---------------------------------------------------------------------------
// gfx1250 async global->LDS copy (ASYNCcnt path) with plain-copy fallback
// ---------------------------------------------------------------------------
#if defined(__has_builtin)
#  if __has_builtin(__builtin_amdgcn_global_load_async_to_lds_b128)
#    define USE_ASYNC_LDS 1
#  endif
#endif
#ifndef USE_ASYNC_LDS
#  define USE_ASYNC_LDS 0
#endif

__device__ __forceinline__ void copy16_to_lds(const void* g, void* l) {
#if USE_ASYNC_LDS
    __builtin_amdgcn_global_load_async_to_lds_b128(
        (__attribute__((address_space(1))) v4i*)(void*)g,
        (__attribute__((address_space(3))) v4i*)l, 0, 0);
#else
    *(uint4*)l = *(const uint4*)g;
#endif
}

__device__ __forceinline__ void lds_copy_wait() {
#if USE_ASYNC_LDS
#  if defined(__has_builtin) && __has_builtin(__builtin_amdgcn_s_wait_asynccnt)
    __builtin_amdgcn_s_wait_asynccnt(0);
#  else
    asm volatile("s_wait_asynccnt 0x0" ::: "memory");
#  endif
#endif
}

__device__ __forceinline__ v8f wmma_bf16(v16bf a, v16bf b, v8f c) {
    // D = A(16x32 bf16) * B(32x16 bf16) + C(16x16 f32)
    return __builtin_amdgcn_wmma_f32_16x16x32_bf16(
        false, a, false, b, (short)0, c, false, false);
}

// ---------------------------------------------------------------------------
// Kernel 1: convert Wk, Wv (f32, [D, E] row-major) to bf16
// ---------------------------------------------------------------------------
__global__ void __launch_bounds__(256)
cvt_bf16_kernel(const float* __restrict__ wk, const float* __restrict__ wv,
                bf16* __restrict__ wkb, bf16* __restrict__ wvb, int n) {
    int i = blockIdx.x * 256 + threadIdx.x;
    if (i < n) {
        wkb[i] = (bf16)wk[i];
        wvb[i] = (bf16)wv[i];
    }
}

// ---------------------------------------------------------------------------
// Kernel 2: hash n-grams + gather 16 tables -> embB bf16 [NTOK, 1024]
// 4 tokens per block.  Hashes computed once per (token, table) by 64 threads,
// broadcast via LDS; gather phase uses 64 lanes per token (coalesced 256B).
// ---------------------------------------------------------------------------
__global__ void __launch_bounds__(256)
hash_gather_kernel(const long long* __restrict__ ids,
                   const long long* __restrict__ mults,
                   const float* __restrict__ tables,
                   bf16* __restrict__ embB) {
    __shared__ int idx_sh[4][16];
    const int tid = threadIdx.x;
    const int tokBase = blockIdx.x * 4;

    if (tid < 64) {
        const int tsub = tid >> 4;
        const int tab  = tid & 15;
        const int tok  = tokBase + tsub;
        const int t    = tok & (T_SEQ - 1);

        const unsigned long long m0 = (unsigned long long)mults[0];
        const unsigned long long m1 = (unsigned long long)mults[1];
        const unsigned long long m2 = (unsigned long long)mults[2];

        const unsigned long long i0 = (unsigned long long)ids[tok];
        const unsigned long long i1 = (t >= 1) ? (unsigned long long)ids[tok - 1] : 0ull;
        const unsigned long long i2 = (t >= 2) ? (unsigned long long)ids[tok - 2] : 0ull;

        // int64 wrap-around semantics via uint64 arithmetic
        unsigned long long h = (i0 * m0) ^ (i1 * m1);
        if (tab >= 8) h ^= (i2 * m2);                    // trigram tables 8..15

        const long long prime = (long long)(VOCAB + (tab & 7) * 7);
        long long r = (long long)h % prime;              // python-style floored mod
        if (r < 0) r += prime;
        if (r > VOCAB - 1) r = VOCAB - 1;                // clip
        idx_sh[tsub][tab] = (int)r;
    }
    __syncthreads();

    const int tsub = tid >> 6;
    const int lane = tid & 63;
    const int tok  = tokBase + tsub;
    bf16* dst = embB + (size_t)tok * E_DIM;

    #pragma unroll
    for (int j = 0; j < 16; ++j) {
        const size_t r = (size_t)idx_sh[tsub][j];
        dst[j * HEAD_DIM + lane] =
            (bf16)tables[((size_t)j * VOCAB + r) * HEAD_DIM + lane];
    }
}

// ---------------------------------------------------------------------------
// Kernel 3: key GEMM fused with gate reduction.
// Block = 256 threads (8 waves), M-tile = 16 tokens, N = 4096 in 16-wide chunks
// across waves, K = 1024.  key is never written to memory.  hidden is read from
// HBM exactly once: the coalesced pass accumulates S2 and caches h*wk_w*wq_w as
// bf16 in LDS for the S3 epilogue.  A-tile staged via async global->LDS.
// ---------------------------------------------------------------------------
__global__ void __launch_bounds__(256)
key_gate_kernel(const bf16* __restrict__ embB, const bf16* __restrict__ WkB,
                const float* __restrict__ bk, const float* __restrict__ hidden,
                const float* __restrict__ wkn, const float* __restrict__ wqn,
                float* __restrict__ gate) {
    __shared__ __align__(16) bf16 Atile[16 * E_DIM];   //  32 KB
    __shared__ __align__(16) bf16 Htile[16 * D_HID];   // 128 KB: h*wk_w*wq_w
    __shared__ float s1_sh[16], s2_sh[16], s3_sh[16];

    const int tid  = threadIdx.x;
    const int tok0 = blockIdx.x * 16;

    if (tid < 16) { s1_sh[tid] = 0.f; s2_sh[tid] = 0.f; s3_sh[tid] = 0.f; }

    // A-tile: 16 tokens x 1024 bf16 (async copy overlaps the hidden pass)
    {
        const uint4* src = (const uint4*)(embB + (size_t)tok0 * E_DIM);
        uint4* dst = (uint4*)Atile;
        #pragma unroll
        for (int i = tid; i < (16 * E_DIM) / 8; i += 256)
            copy16_to_lds(src + i, dst + i);
    }

    // single coalesced hidden pass: S2 partials + weighted-hidden LDS tile
    {
        float acc = 0.f;
        int curm = 0;
        const float* hb = hidden + (size_t)tok0 * D_HID;
        for (int i = tid; i < 16 * D_HID; i += 256) {
            const int m = i >> 12;
            const int d = i & (D_HID - 1);
            if (m != curm) { atomicAdd(&s2_sh[curm], acc); acc = 0.f; curm = m; }
            const float h = hb[i];
            acc += h * h;
            Htile[i] = (bf16)(h * wkn[d] * wqn[d]);
        }
        atomicAdd(&s2_sh[curm], acc);
    }
    lds_copy_wait();
    __syncthreads();

    const int wave = tid >> 5;
    const int lane = tid & 31;
    const int half = lane >> 4;   // 0: lanes 0-15, 1: lanes 16-31
    const int lmod = lane & 15;

    float s1a[8] = {0.f, 0.f, 0.f, 0.f, 0.f, 0.f, 0.f, 0.f};
    float s3a[8] = {0.f, 0.f, 0.f, 0.f, 0.f, 0.f, 0.f, 0.f};

    // A 16-bit layout: lanes 0-15 read K[0..7],[16..23]; lanes 16-31 K[8..15],[24..31]
    const bf16* Arow = Atile + (size_t)lmod * E_DIM + half * 8;

    for (int iter = 0; iter < ITERS; ++iter) {           // uniform loop counter
        const int chunk = wave + iter * 8;
        const int n0 = chunk * 16;
        v8f c = {};
        const bf16* Brow  = WkB + (size_t)(n0 + lmod) * E_DIM + half * 8;
        const bf16* Bnext = Brow + (size_t)128 * E_DIM;  // this wave's next chunk
        const bool pf = (iter + 1) < ITERS;              // uniform -> scalar branch
        for (int kk = 0; kk < E_DIM; kk += 32) {
            if (pf && (kk & 63) == 0)
                __builtin_prefetch(Bnext + kk, 0, 1);    // global_prefetch_b8
            FragAB a, b;
            const uint4* ap = (const uint4*)(Arow + kk);
            a.q[0] = ap[0];
            a.q[1] = ap[2];   // +16 elements
            const uint4* bp = (const uint4*)(Brow + kk);
            b.q[0] = bp[0];
            b.q[1] = bp[2];
            c = wmma_bf16(a.v, b.v, c);
        }
        // epilogue: lane holds key[m = r + half*8][n0 + lmod] in c[r]
        const float bkv = bk[n0 + lmod];
        #pragma unroll
        for (int r = 0; r < 8; ++r) {
            const int m = r + half * 8;
            const float key = c[r] + bkv;
            const float hw  = (float)Htile[m * D_HID + n0 + lmod];
            s1a[r] += key * key;
            s3a[r] += key * hw;
        }
    }

    // reduce across the 16 lanes of each half-wave, then across waves via LDS
    #pragma unroll
    for (int r = 0; r < 8; ++r) {
        float a1 = s1a[r], a3 = s3a[r];
        for (int off = 8; off; off >>= 1) {
            a1 += __shfl_xor(a1, off, 16);
            a3 += __shfl_xor(a3, off, 16);
        }
        if (lmod == 0) {
            atomicAdd(&s1_sh[r + half * 8], a1);
            atomicAdd(&s3_sh[r + half * 8], a3);
        }
    }
    __syncthreads();

    if (tid < 16) {
        const float mk = s1_sh[tid] * (1.f / D_HID) + EPS_F;
        const float mh = s2_sh[tid] * (1.f / D_HID) + EPS_F;
        float g = s3_sh[tid] * rsqrtf(mk) * rsqrtf(mh) * (1.f / 64.f); // /sqrt(D)
        const float sg = (g > 0.f) ? 1.f : ((g < 0.f) ? -1.f : 0.f);
        const float s = sqrtf(fmaxf(fabsf(g), 1e-6f)) * sg;
        gate[tok0 + tid] = 1.f / (1.f + __expf(-s));
    }
}

// ---------------------------------------------------------------------------
// Kernel 4 (fused): value GEMM + gate + causal depthwise conv4 + SiLU + add.
// Block owns 32 output tokens but computes a 48-row tile (16-row halo) so the
// conv taps t-1..t-3 are available in-register.  The 48-value column per (d)
// is assembled with one half-wave __shfl_xor(16) (C-layout: M=r in lanes 0-15,
// M=8+r in lanes 16-31 at the same N).  value never touches HBM.
// ---------------------------------------------------------------------------
__global__ void __launch_bounds__(256)
value_conv_kernel(const bf16* __restrict__ embB, const bf16* __restrict__ WvB,
                  const float* __restrict__ bv, const float* __restrict__ gate,
                  const float* __restrict__ convw, float* __restrict__ out) {
    __shared__ __align__(16) bf16 Atile[MTILE * E_DIM];   // 96 KB

    const int tid     = threadIdx.x;
    const int tokOut0 = blockIdx.x * MOUT;
    const int tokH    = tokOut0 - 16;      // first halo row (may be < 0)

    // A-tile load (48 rows, halo rows clamped; masked later) via async copy
    {
        const uint4* src = (const uint4*)embB;
        uint4* dst = (uint4*)Atile;
        for (int i = tid; i < (MTILE * E_DIM) / 8; i += 256) {
            const int row = i >> 7;          // 128 uint4 per row
            const int col = i & 127;
            int stok = tokH + row;
            if (stok < 0) stok = 0;
            copy16_to_lds(src + (size_t)stok * 128 + col, dst + i);
        }
    }

    const int wave = tid >> 5;
    const int lane = tid & 31;
    const int half = lane >> 4;
    const int lmod = lane & 15;

    float garr[3][8];
    #pragma unroll
    for (int tau = 0; tau < 3; ++tau)
        #pragma unroll
        for (int r = 0; r < 8; ++r) {
            int tk = tokH + tau * 16 + half * 8 + r;
            if (tk < 0) tk = 0;
            garr[tau][r] = gate[tk];
        }

    lds_copy_wait();
    __syncthreads();

    const bf16* A0 = Atile + (size_t)lmod * E_DIM + half * 8;
    const bf16* A1 = A0 + (size_t)16 * E_DIM;
    const bf16* A2 = A0 + (size_t)32 * E_DIM;

    for (int iter = 0; iter < ITERS; ++iter) {           // uniform loop counter
        const int chunk = wave + iter * 8;
        const int n0 = chunk * 16;
        const int d  = n0 + lmod;
        v8f c0 = {}, c1 = {}, c2 = {};
        const bf16* Brow  = WvB + (size_t)d * E_DIM + half * 8;
        const bf16* Bnext = Brow + (size_t)128 * E_DIM;
        const bool pf = (iter + 1) < ITERS;
        for (int kk = 0; kk < E_DIM; kk += 32) {
            if (pf && (kk & 63) == 0)
                __builtin_prefetch(Bnext + kk, 0, 1);
            FragAB b;
            const uint4* bp = (const uint4*)(Brow + kk);
            b.q[0] = bp[0];
            b.q[1] = bp[2];
            FragAB a0, a1, a2;
            const uint4* ap0 = (const uint4*)(A0 + kk);
            const uint4* ap1 = (const uint4*)(A1 + kk);
            const uint4* ap2 = (const uint4*)(A2 + kk);
            a0.q[0] = ap0[0]; a0.q[1] = ap0[2];
            a1.q[0] = ap1[0]; a1.q[1] = ap1[2];
            a2.q[0] = ap2[0]; a2.q[1] = ap2[2];
            c0 = wmma_bf16(a0.v, b.v, c0);
            c1 = wmma_bf16(a1.v, b.v, c1);
            c2 = wmma_bf16(a2.v, b.v, c2);
        }

        const float bvv = bv[d];
        float colv[MTILE];
        #pragma unroll
        for (int r = 0; r < 8; ++r) {
            float o0 = garr[0][r] * (c0[r] + bvv);
            float o1 = garr[1][r] * (c1[r] + bvv);
            float o2 = garr[2][r] * (c2[r] + bvv);
            float x0 = __shfl_xor(o0, 16);
            float x1 = __shfl_xor(o1, 16);
            float x2 = __shfl_xor(o2, 16);
            colv[ 0 + half * 8 + r] = o0;
            colv[ 0 + (half ^ 1) * 8 + r] = x0;
            colv[16 + half * 8 + r] = o1;
            colv[16 + (half ^ 1) * 8 + r] = x1;
            colv[32 + half * 8 + r] = o2;
            colv[32 + (half ^ 1) * 8 + r] = x2;
        }

        const float w0 = convw[d * 4 + 0];
        const float w1 = convw[d * 4 + 1];
        const float w2 = convw[d * 4 + 2];
        const float w3 = convw[d * 4 + 3];

        #pragma unroll
        for (int m = 16; m < MTILE; ++m) {
            const int tok = tokOut0 + m - 16;
            const int t   = tok & (T_SEQ - 1);
            const float cur = colv[m];
            const float x0 = (t >= 3) ? colv[m - 3] : 0.f;
            const float x1 = (t >= 2) ? colv[m - 2] : 0.f;
            const float x2 = (t >= 1) ? colv[m - 1] : 0.f;
            const float cv = x0 * w0 + x1 * w1 + x2 * w2 + cur * w3;
            const float si = cv / (1.f + __expf(-cv));
            out[(size_t)tok * D_HID + d] = cur + si;
        }
    }
}

// ---------------------------------------------------------------------------
extern "C" void kernel_launch(void* const* d_in, const int* in_sizes, int n_in,
                              void* d_out, int out_size, void* d_ws, size_t ws_size,
                              hipStream_t stream) {
    const float*     hidden = (const float*)d_in[0];
    const long long* ids    = (const long long*)d_in[1];
    const long long* mults  = (const long long*)d_in[2];
    const float*     tables = (const float*)d_in[3];
    const float*     Wk     = (const float*)d_in[4];
    const float*     bk     = (const float*)d_in[5];
    const float*     Wv     = (const float*)d_in[6];
    const float*     bv     = (const float*)d_in[7];
    const float*     wkn    = (const float*)d_in[8];
    const float*     wqn    = (const float*)d_in[9];
    const float*     convw  = (const float*)d_in[10];
    float* out = (float*)d_out;

    char* ws = (char*)d_ws;
    bf16*  embB    = (bf16*)(ws);              // 16,777,216 B
    bf16*  WkB     = (bf16*)(ws + 16777216);   //  8,388,608 B
    bf16*  WvB     = (bf16*)(ws + 25165824);   //  8,388,608 B
    float* gateBuf = (float*)(ws + 33554432);  //     32,768 B

    cvt_bf16_kernel<<<(D_HID * E_DIM + 255) / 256, 256, 0, stream>>>(
        Wk, Wv, WkB, WvB, D_HID * E_DIM);

    hash_gather_kernel<<<NTOK / 4, 256, 0, stream>>>(ids, mults, tables, embB);

    key_gate_kernel<<<NTOK / 16, 256, 0, stream>>>(
        embB, WkB, bk, hidden, wkn, wqn, gateBuf);

    value_conv_kernel<<<NTOK / MOUT, 256, 0, stream>>>(
        embB, WvB, bv, gateBuf, convw, out);
}
